// SelfAttention_28982439313447
// MI455X (gfx1250) — matrix operand order repaired and checked by reading, not compiled, not run on hardware
//
#include <hip/hip_runtime.h>
#include <hip/hip_bf16.h>

// ---------------------------------------------------------------------------
// Self-attention, CDNA5 (gfx1250) wave32 WMMA bf16 pipeline + TDM K-staging.
//   B=4, S=2048, D=1024, H=8, Hd=128
// ---------------------------------------------------------------------------

typedef __attribute__((ext_vector_type(16))) __bf16        v16bf;
typedef __attribute__((ext_vector_type(8)))  float         v8f;
typedef __attribute__((ext_vector_type(4)))  unsigned int  u32x4;
typedef __attribute__((ext_vector_type(8)))  int           i32x8;
typedef __attribute__((ext_vector_type(4)))  int           i32x4;

#define EMBED   1024
#define SEQ     2048
#define BATCH   4
#define HEADS   8
#define HDIM    128
#define N3      (3 * EMBED)
#define ROWS    (BATCH * SEQ)          // 8192
#define LOG2E   1.4426950408889634f

#if __has_builtin(__builtin_amdgcn_tensor_load_to_lds)
#define USE_TDM 1
#else
#define USE_TDM 0
#endif
#if USE_TDM && __has_include(<hip/amd_detail/amd_gfx1250_TDM.h>)
#define TDM_SIX_ARG 1          // therock clang-23 headers present -> 6-arg builtin
#else
#define TDM_SIX_ARG 0          // ROCm 7.2 clang-22 -> 5-arg builtin
#endif

union ABFrag { uint4 u[2]; v16bf v; };

// A fragment (16x32 bf16) from row-major A[lda] per ISA layout:
//   lane L: row = m0+(L&15); VGPR j: K = 8*(L>>4)+2j,+1 ; VGPR 4+j: K = 16+8*(L>>4)+2j,+1
__device__ __forceinline__ v16bf load_a16x32(const __bf16* A, int lda, int m0, int kb) {
    int lane = threadIdx.x & 31;
    const __bf16* p = A + (size_t)(m0 + (lane & 15)) * lda + kb + ((lane >> 4) << 3);
    ABFrag f;
    f.u[0] = *(const uint4*)p;
    f.u[1] = *(const uint4*)(p + 16);
    return f.v;
}

// B fragment (32x16 bf16) from TRANSPOSED storage Bt[N][K]:
//   lane L: col = n0+(L&15); K = kb + 8*(L>>4) + 0..7, kb + 16 + 8*(L>>4) + 0..7
__device__ __forceinline__ v16bf load_b32x16_t(const __bf16* Bt, int ldb, int n0, int kb) {
    int lane = threadIdx.x & 31;
    const __bf16* p = Bt + (size_t)(n0 + (lane & 15)) * ldb + kb + ((lane >> 4) << 3);
    ABFrag f;
    f.u[0] = *(const uint4*)p;
    f.u[1] = *(const uint4*)(p + 16);
    return f.v;
}

// Same B fragment but from an LDS K-slab laid out [32][HDIM]
__device__ __forceinline__ v16bf load_b32x16_lds(const __bf16* slab, int n0, int k0) {
    int lane = threadIdx.x & 31;
    const __bf16* p = slab + (size_t)(n0 + (lane & 15)) * HDIM + k0 + ((lane >> 4) << 3);
    ABFrag f;
    f.u[0] = *(const uint4*)p;
    f.u[1] = *(const uint4*)(p + 16);
    return f.v;
}

__device__ __forceinline__ v8f wmma_bf16(v16bf a, v16bf b, v8f c) {
    return __builtin_amdgcn_wmma_f32_16x16x32_bf16(false, a, false, b, (short)0, c,
                                                   false, false);
}

// ---------------------------------------------------------------------------
// TDM: async-load one 32x128 bf16 K-block (row-major, stride HDIM) into LDS.
// D# built per CDNA5 ISA.
__device__ __forceinline__ void issue_k_slab(const __bf16* gsrc, __bf16* slab) {
    // make sure our own ds_reads of this slab from the previous round retired
    asm volatile("s_wait_dscnt 0x0" ::: "memory");
#if USE_TDM
    unsigned lds_off = (unsigned)(size_t)slab;                 // flat addr[31:0] == LDS offset
    unsigned long long ga = (unsigned long long)(size_t)gsrc;  // byte address of tile start
    u32x4 g0;
    g0[0] = 1u;                                                // count=1 (valid user D#)
    g0[1] = lds_off;                                           // lds_addr
    g0[2] = (unsigned)(ga & 0xffffffffu);                      // global_addr[31:0]
    g0[3] = (unsigned)((ga >> 32) & 0x1ffffffu) | (2u << 30);  // global_addr[56:32] | type=2
    i32x8 g1;
    g1[0] = (int)(1u << 16);       // workgroup_mask=0, data_size=1 (2 bytes)
    g1[1] = (int)(HDIM << 16);     // tensor_dim0 = 128 (bits 79:48, low half)
    g1[2] = (int)(32u << 16);      // tensor_dim0 hi=0 | tensor_dim1 = 32
    g1[3] = (int)(HDIM << 16);     // tensor_dim1 hi=0 | tile_dim0 = 128
    g1[4] = 32;                    // tile_dim1 = 32, tile_dim2 = 0
    g1[5] = HDIM;                  // tensor_dim0_stride = 128 (lo32)
    g1[6] = 0;                     // stride hi | tensor_dim1_stride lo
    g1[7] = 0;
    i32x4 z4 = {0, 0, 0, 0};
#if TDM_SIX_ARG
    i32x8 z8 = {0, 0, 0, 0, 0, 0, 0, 0};
    __builtin_amdgcn_tensor_load_to_lds(g0, g1, z4, z4, z8, 0);
#else
    __builtin_amdgcn_tensor_load_to_lds(g0, g1, z4, z4, 0);
#endif
#else
    // Fallback: cooperative register copy, 8 KB per wave = 16 x b128 per lane
    int lane = threadIdx.x & 31;
    const uint4* s = (const uint4*)gsrc;
    uint4* d = (uint4*)slab;
#pragma unroll
    for (int j = 0; j < 16; j++) d[lane + 32 * j] = s[lane + 32 * j];
#endif
}

__device__ __forceinline__ void wait_k_slab(bool pending) {
#if USE_TDM
#if __has_builtin(__builtin_amdgcn_s_wait_tensorcnt)
    if (pending) __builtin_amdgcn_s_wait_tensorcnt(1);
    else         __builtin_amdgcn_s_wait_tensorcnt(0);
#else
    if (pending) asm volatile("s_wait_tensorcnt 0x1" ::: "memory");
    else         asm volatile("s_wait_tensorcnt 0x0" ::: "memory");
#endif
#else
    (void)pending;
    asm volatile("s_wait_dscnt 0x0" ::: "memory");
#endif
}

// ---------------------------------------------------------------------------
// Elementwise fp32 -> bf16
__global__ void cvt_f32_bf16(const float* __restrict__ in, __bf16* __restrict__ out, int n) {
    int i = blockIdx.x * blockDim.x + threadIdx.x;
    if (i < n) out[i] = (__bf16)in[i];
}

// in [rows][cols] fp32 -> out [cols][rows] bf16
__global__ void transpose_cvt(const float* __restrict__ in, __bf16* __restrict__ out,
                              int rows, int cols) {
    int i = blockIdx.x * blockDim.x + threadIdx.x;
    if (i < rows * cols) {
        int r = i / cols, c = i % cols;
        out[(size_t)c * rows + r] = (__bf16)in[i];
    }
}

// ---------------------------------------------------------------------------
// Double-buffered 32x64 GEMM tile helpers
__device__ __forceinline__ void load_ab(const __bf16* A, const __bf16* Bt, int lda, int ldb,
                                        int m0, int c0, int kb, v16bf a[2], v16bf b[4]) {
    a[0] = load_a16x32(A, lda, m0, kb);
    a[1] = load_a16x32(A, lda, m0 + 16, kb);
#pragma unroll
    for (int t = 0; t < 4; t++) b[t] = load_b32x16_t(Bt, ldb, c0 + t * 16, kb);
}

__device__ __forceinline__ void do_wmma8(const v16bf a[2], const v16bf b[4], v8f acc[2][4]) {
#pragma unroll
    for (int t = 0; t < 4; t++) {
        acc[0][t] = wmma_bf16(a[0], b[t], acc[0][t]);
        acc[1][t] = wmma_bf16(a[1], b[t], acc[1][t]);
    }
}

// ---------------------------------------------------------------------------
// QKV GEMM: [8192 x 1024] x [1024 x 3072]^T-stored -> scatter Q[S][Hd], K[S][Hd],
// Vt[Hd][S] bf16 per batch-head. One wave = 32 rows x 64 cols, double buffered.
// waves_per_eu(1): allow full VGPR budget -> no scratch spills in the hot loop.
__global__ void __launch_bounds__(256)
__attribute__((amdgpu_waves_per_eu(1)))
qkv_gemm(const __bf16* __restrict__ X, const __bf16* __restrict__ Wt,
         const float* __restrict__ bias,
         __bf16* __restrict__ Qb, __bf16* __restrict__ Kb,
         __bf16* __restrict__ Vt) {
    int wave = (int)((blockIdx.x * blockDim.x + threadIdx.x) >> 5);
    int lane = threadIdx.x & 31;
    const int colGroups = N3 / 64;                 // 48
    int m0 = (wave / colGroups) * 32;
    int c0 = (wave % colGroups) * 64;

    v8f acc[2][4] = {};
    v16bf a0[2], b0[4], a1[2], b1[4];
    load_ab(X, Wt, EMBED, EMBED, m0, c0, 0, a0, b0);
    for (int kb = 0; kb < EMBED; kb += 64) {
        load_ab(X, Wt, EMBED, EMBED, m0, c0, kb + 32, a1, b1);
        do_wmma8(a0, b0, acc);
        if (kb + 64 < EMBED) load_ab(X, Wt, EMBED, EMBED, m0, c0, kb + 64, a0, b0);
        do_wmma8(a1, b1, acc);
    }

    int nloc = lane & 15, halfId = lane >> 4;
#pragma unroll
    for (int rt = 0; rt < 2; rt++) {
#pragma unroll
        for (int t = 0; t < 4; t++) {
            int col  = c0 + t * 16 + nloc;
            float bv = bias[col];
            int part = col >> 10;                  // 0=q 1=k 2=v
            int d    = col & 1023;
            int h    = d >> 7;
            int hd   = d & 127;
#pragma unroll
            for (int r = 0; r < 8; r++) {
                int g  = m0 + rt * 16 + r + halfId * 8;
                int b_ = g >> 11;
                int s  = g & (SEQ - 1);
                __bf16 val = (__bf16)(acc[rt][t][r] + bv);
                size_t bh = (size_t)(b_ * HEADS + h);
                if (part == 0)      Qb[(bh * SEQ + s) * HDIM + hd] = val;
                else if (part == 1) Kb[(bh * SEQ + s) * HDIM + hd] = val;
                else                Vt[(bh * HDIM + hd) * SEQ + s] = val;
            }
        }
    }
}

// ---------------------------------------------------------------------------
// One 32-key flash-attention step for a 16-query tile (K block already in LDS).
__device__ __forceinline__ void attn_kblock(const __bf16* Ksl, const __bf16* V, int kb,
                                            const v16bf qf[4], v8f acc[8],
                                            float mrow[8], float lrow[8],
                                            __bf16 (*psh)[32]) {
    const float scale = 0.08838834764831845f;      // Hd^-0.5
    int lane = threadIdx.x & 31;
    int nloc = lane & 15, halfId = lane >> 4;

    // first half of the V fragments early: latency hides under score WMMA+softmax
    v16bf vf[4];
#pragma unroll
    for (int t = 0; t < 4; t++) vf[t] = load_b32x16_t(V, SEQ, t * 16, kb);

    // ---- scores from LDS K-slab ----
    v16bf kf[4];
    v8f s0 = {}, s1 = {};
#pragma unroll
    for (int i = 0; i < 4; i++) kf[i] = load_b32x16_lds(Ksl, 0, i * 32);
#pragma unroll
    for (int i = 0; i < 4; i++) s0 = wmma_bf16(qf[i], kf[i], s0);
#pragma unroll
    for (int i = 0; i < 4; i++) kf[i] = load_b32x16_lds(Ksl, 16, i * 32);
#pragma unroll
    for (int i = 0; i < 4; i++) s1 = wmma_bf16(qf[i], kf[i], s1);

    // ---- online softmax (row r+8*halfId lives across a 16-lane group) ----
    float p0[8], p1[8], corr[8];
#pragma unroll
    for (int r = 0; r < 8; r++) {
        float t0 = s0[r] * scale, t1 = s1[r] * scale;
        float mx = fmaxf(t0, t1);
#pragma unroll
        for (int off = 1; off < 16; off <<= 1)
            mx = fmaxf(mx, __shfl_xor(mx, off, 32));
        float mnew = fmaxf(mrow[r], mx);
        float c  = __builtin_exp2f((mrow[r] - mnew) * LOG2E);
        float e0 = __builtin_exp2f((t0 - mnew) * LOG2E);
        float e1 = __builtin_exp2f((t1 - mnew) * LOG2E);
        float rs = e0 + e1;
#pragma unroll
        for (int off = 1; off < 16; off <<= 1)
            rs += __shfl_xor(rs, off, 32);
        lrow[r] = lrow[r] * c + rs;
        mrow[r] = mnew;
        corr[r] = c;
        p0[r] = e0; p1[r] = e1;
    }
#pragma unroll
    for (int t = 0; t < 8; t++)
#pragma unroll
        for (int r = 0; r < 8; r++)
            acc[t][r] *= corr[r];

    // ---- C-layout -> A-layout via per-wave LDS tile ----
#pragma unroll
    for (int r = 0; r < 8; r++) {
        psh[r + halfId * 8][nloc]      = (__bf16)p0[r];
        psh[r + halfId * 8][16 + nloc] = (__bf16)p1[r];
    }
    asm volatile("" ::: "memory");
    asm volatile("s_wait_dscnt 0x0" ::: "memory");
    ABFrag pf;
    {
        int row = lane & 15;
        int h8  = halfId * 8;
        pf.u[0] = *(const uint4*)&psh[row][h8];
        pf.u[1] = *(const uint4*)&psh[row][16 + h8];
    }
    asm volatile("" ::: "memory");

    // ---- out += P @ V_block ----
#pragma unroll
    for (int t = 0; t < 4; t++) acc[t] = wmma_bf16(pf.v, vf[t], acc[t]);
#pragma unroll
    for (int t = 0; t < 4; t++) vf[t] = load_b32x16_t(V, SEQ, (t + 4) * 16, kb);
#pragma unroll
    for (int t = 0; t < 4; t++) acc[t + 4] = wmma_bf16(pf.v, vf[t], acc[t + 4]);
}

// Flash attention: 2 waves/block; each wave owns a 16-query tile of one
// batch-head, K blocks staged into LDS by the Tensor Data Mover (double
// buffered, s_wait_tensorcnt-synchronized).
__global__ void __launch_bounds__(64)
__attribute__((amdgpu_waves_per_eu(1)))
flash_attn(const __bf16* __restrict__ Qb,
           const __bf16* __restrict__ Kb,
           const __bf16* __restrict__ Vt,
           __bf16* __restrict__ O) {
    __shared__ __align__(16) __bf16 ksh[2][2][32][HDIM];   // 32 KB
    __shared__ __align__(16) __bf16 psh[2][16][32];        //  2 KB

    int wslot = threadIdx.x >> 5;
    int lane  = threadIdx.x & 31;
    int wave  = (int)blockIdx.x * 2 + wslot;               // 0..4095
    int qt = wave & 127;
    int bh = wave >> 7;
    const __bf16* Q = Qb + (size_t)bh * SEQ * HDIM;
    const __bf16* K = Kb + (size_t)bh * SEQ * HDIM;
    const __bf16* V = Vt + (size_t)bh * HDIM * SEQ;        // [Hd][S]
    int m0 = qt * 16;

    v16bf qf[4];
#pragma unroll
    for (int i = 0; i < 4; i++) qf[i] = load_a16x32(Q, HDIM, m0, i * 32);

    v8f acc[8] = {};
    float mrow[8], lrow[8];
#pragma unroll
    for (int r = 0; r < 8; r++) { mrow[r] = -3.0e38f; lrow[r] = 0.f; }

    issue_k_slab(K, &ksh[wslot][0][0][0]);
    for (int kb = 0; kb < SEQ; kb += 64) {
        issue_k_slab(K + (size_t)(kb + 32) * HDIM, &ksh[wslot][1][0][0]);
        wait_k_slab(true);
        attn_kblock(&ksh[wslot][0][0][0], V, kb, qf, acc, mrow, lrow, psh[wslot]);
        bool more = (kb + 64) < SEQ;
        if (more) issue_k_slab(K + (size_t)(kb + 64) * HDIM, &ksh[wslot][0][0][0]);
        wait_k_slab(more);
        attn_kblock(&ksh[wslot][1][0][0], V, kb + 32, qf, acc, mrow, lrow, psh[wslot]);
    }

    // ---- epilogue: normalize, store bf16 to merged-head [B][S][D] ----
    int b_ = bh >> 3, h = bh & 7;
    int nloc = lane & 15, halfId = lane >> 4;
#pragma unroll
    for (int t = 0; t < 8; t++) {
        int col = h * HDIM + t * 16 + nloc;
#pragma unroll
        for (int r = 0; r < 8; r++) {
            int row = m0 + r + halfId * 8;
            O[((size_t)b_ * SEQ + row) * EMBED + col] = (__bf16)(acc[t][r] / lrow[r]);
        }
    }
}

// ---------------------------------------------------------------------------
// Output projection: [8192 x 1024] x [1024 x 1024] + bias -> fp32 d_out
__global__ void __launch_bounds__(256)
__attribute__((amdgpu_waves_per_eu(1)))
proj_gemm(const __bf16* __restrict__ A, const __bf16* __restrict__ Wt,
          const float* __restrict__ bias, float* __restrict__ Out) {
    int wave = (int)((blockIdx.x * blockDim.x + threadIdx.x) >> 5);
    int lane = threadIdx.x & 31;
    const int colGroups = EMBED / 64;              // 16
    int m0 = (wave / colGroups) * 32;
    int c0 = (wave % colGroups) * 64;

    v8f acc[2][4] = {};
    v16bf a0[2], b0[4], a1[2], b1[4];
    load_ab(A, Wt, EMBED, EMBED, m0, c0, 0, a0, b0);
    for (int kb = 0; kb < EMBED; kb += 64) {
        load_ab(A, Wt, EMBED, EMBED, m0, c0, kb + 32, a1, b1);
        do_wmma8(a0, b0, acc);
        if (kb + 64 < EMBED) load_ab(A, Wt, EMBED, EMBED, m0, c0, kb + 64, a0, b0);
        do_wmma8(a1, b1, acc);
    }

    int nloc = lane & 15, halfId = lane >> 4;
#pragma unroll
    for (int rt = 0; rt < 2; rt++) {
#pragma unroll
        for (int t = 0; t < 4; t++) {
            int col  = c0 + t * 16 + nloc;
            float bv = bias[col];
#pragma unroll
            for (int r = 0; r < 8; r++) {
                int row = m0 + rt * 16 + r + halfId * 8;
                Out[(size_t)row * EMBED + col] = acc[rt][t][r] + bv;
            }
        }
    }
}

// ---------------------------------------------------------------------------
extern "C" void kernel_launch(void* const* d_in, const int* in_sizes, int n_in,
                              void* d_out, int out_size, void* d_ws, size_t ws_size,
                              hipStream_t stream) {
    const float* X     = (const float*)d_in[0];   // [B,S,D]
    const float* Wqkv  = (const float*)d_in[1];   // [D, 3D]
    const float* Bqkv  = (const float*)d_in[2];   // [3D]
    const float* Wproj = (const float*)d_in[3];   // [D, D]
    const float* Bproj = (const float*)d_in[4];   // [D]
    float* Out = (float*)d_out;

    char* ws = (char*)d_ws;
    size_t off = 0;
    __bf16* Xbf   = (__bf16*)(ws + off); off += (size_t)ROWS * EMBED * 2;      // 16 MB
    __bf16* Wqt   = (__bf16*)(ws + off); off += (size_t)N3 * EMBED * 2;        //  6 MB
    __bf16* Wpt   = (__bf16*)(ws + off); off += (size_t)EMBED * EMBED * 2;     //  2 MB
    __bf16* Qb    = (__bf16*)(ws + off); off += (size_t)ROWS * EMBED * 2;      // 16 MB
    __bf16* Kb    = (__bf16*)(ws + off); off += (size_t)ROWS * EMBED * 2;      // 16 MB
    __bf16* Vt    = (__bf16*)(ws + off); off += (size_t)ROWS * EMBED * 2;      // 16 MB
    __bf16* Attn  = (__bf16*)(ws + off); off += (size_t)ROWS * EMBED * 2;      // 16 MB
    (void)ws_size; (void)in_sizes; (void)n_in; (void)out_size;

    {
        int n = ROWS * EMBED;                                   // 8388608
        cvt_f32_bf16<<<n / 256, 256, 0, stream>>>(X, Xbf, n);
    }
    {
        int n = EMBED * N3;                                     // 3145728
        transpose_cvt<<<(n + 255) / 256, 256, 0, stream>>>(Wqkv, Wqt, EMBED, N3);
    }
    {
        int n = EMBED * EMBED;                                  // 1048576
        transpose_cvt<<<(n + 255) / 256, 256, 0, stream>>>(Wproj, Wpt, EMBED, EMBED);
    }

    // QKV projection: 256 row-tiles x 48 col-groups = 12288 waves -> 1536 blocks
    qkv_gemm<<<1536, 256, 0, stream>>>(Xbf, Wqt, Bqkv, Qb, Kb, Vt);

    // Flash attention: 32 bh x 128 q-tiles = 4096 waves -> 2048 blocks x 64
    flash_attn<<<2048, 64, 0, stream>>>(Qb, Kb, Vt, Attn);

    // Output projection: 256 x 16 = 4096 waves -> 512 blocks
    proj_gemm<<<512, 256, 0, stream>>>(Attn, Wpt, Bproj, Out);
}
